// FeatureExtractor_34823594836532
// MI455X (gfx1250) — matrix operand order repaired and checked by reading, not compiled, not run on hardware
//
#include <hip/hip_runtime.h>

typedef __attribute__((ext_vector_type(16))) _Float16 v16h;
typedef __attribute__((ext_vector_type(8)))  _Float16 v8h;
typedef __attribute__((ext_vector_type(8)))  float    v8f;
typedef __attribute__((ext_vector_type(4)))  float    v4f;

namespace {
constexpr int T_STEPS = 60;
constexpr int B_DIM   = 16;
constexpr int S_DIM   = 2000;
constexpr int C_DIM   = 64;
constexpr int H_DIM   = 64;
constexpr int WPAD    = 72;           // padded LDS row in halves (144B, 16B aligned)
constexpr int WAVES   = 4;            // one wave per SIMD in the WGP
constexpr int NTILES  = (B_DIM * S_DIM) / 16;   // 2000 sixteen-sequence tiles
}

__device__ __forceinline__ float fast_sigmoid(float x) {
    return 1.0f / (1.0f + __expf(-x));
}
__device__ __forceinline__ float fast_tanh(float x) {
    // tanh(x) = 1 - 2/(exp(2x)+1); saturates correctly at exp under/overflow
    return 1.0f - 2.0f / (1.0f + __expf(2.0f * x));
}

// Load one WMMA B-tile (32x16, K x N) lane-slice straight from f32 row-major W:
// lane = output column (row of W); K runs [koff,koff+8) and [koff+16,koff+24)
// (koff already includes the lanes[16..31] -> K+8 shift). 4x b128 + cvt.
__device__ __forceinline__ v16h load_w_tile_f32(const float* __restrict__ W,
                                                int row, int koff) {
    const float* p = W + row * 64 + koff;
    v4f f0 = *(const v4f*)(p);
    v4f f1 = *(const v4f*)(p + 4);
    v4f f2 = *(const v4f*)(p + 16);
    v4f f3 = *(const v4f*)(p + 20);
    v16h r;
#pragma unroll
    for (int e = 0; e < 4; ++e) {
        r[e]      = (_Float16)f0[e];
        r[4 + e]  = (_Float16)f1[e];
        r[8 + e]  = (_Float16)f2[e];
        r[12 + e] = (_Float16)f3[e];
    }
    return r;
}

__device__ __forceinline__ v8f wmma_f16(v16h a, v16h b, v8f c) {
    // (neg_a, A, neg_b, B, c_mod, C, reuse_a, reuse_b)
    return __builtin_amdgcn_wmma_f32_16x16x32_f16(false, a, false, b, (short)0, c,
                                                  false, false);
}

__global__ __launch_bounds__(WAVES * 32, 1)
void gru_scan_kernel(const float* __restrict__ chars,
                     const float* __restrict__ W_ih,
                     const float* __restrict__ W_hh,
                     const float* __restrict__ b_ih,
                     const float* __restrict__ b_hh,
                     float* __restrict__ out)
{
    // LDS: only the per-wave hidden-state buffer (cross-lane relayout).
    __shared__ __align__(16) _Float16 sH[WAVES][16 * WPAD];   // 9216 B total

    const int tid   = threadIdx.x;
    const int wave  = tid >> 5;
    const int lane  = tid & 31;     // wave32
    const int lrow  = lane & 15;    // A row / B column / D column within tile
    const int lhalf = lane >> 4;    // 0 or 1 (K / M half selector)

    _Float16* hbuf = &sH[wave][0];
    for (int idx = lane; idx < 16 * WPAD; idx += 32) hbuf[idx] = (_Float16)0.0f;

    const int tile = blockIdx.x * WAVES + wave;  // 0..1999
    const int n0   = tile * 16;                  // first sequence of this tile
    const int b    = n0 / S_DIM;                 // S_DIM % 16 == 0 -> no straddle
    const int s0   = n0 - b * S_DIM;

    // ---- Register-resident weights in WMMA B-layout: 48 tiles x 8 VGPRs ----
    // [column-group i][gate r/z/n][K-chunk q]
    v16h Wih[4][3][2], Whh[4][3][2];
#pragma unroll
    for (int i = 0; i < 4; ++i) {
#pragma unroll
        for (int g = 0; g < 3; ++g) {
#pragma unroll
            for (int q = 0; q < 2; ++q) {
                const int row  = g * 64 + i * 16 + lrow;
                const int koff = q * 32 + 8 * lhalf;
                Wih[i][g][q] = load_w_tile_f32(W_ih, row, koff);
                Whh[i][g][q] = load_w_tile_f32(W_hh, row, koff);
            }
        }
    }

    // Per-lane gate biases (depend only on output column = lrow within group).
    float br[4], bz[4], bin[4], bhn[4];
#pragma unroll
    for (int i = 0; i < 4; ++i) {
        const int g = i * 16 + lrow;
        br[i]  = b_ih[g]       + b_hh[g];        // r: fuse both biases
        bz[i]  = b_ih[64 + g]  + b_hh[64 + g];   // z: fuse both biases
        bin[i] = b_ih[128 + g];                  // n: input part separate
        bhn[i] = b_hh[128 + g];                  // n: hidden part separate
    }

    // Hidden state, f32 in D-layout: 4 column groups x 8 accumulator slots.
    v8f hD[4] = {};

    const size_t xstep  = (size_t)S_DIM * C_DIM;  // floats per timestep
    const float* xtile0 = chars + ((size_t)b * T_STEPS * S_DIM + (size_t)s0) * C_DIM;
    const float* xrow0  = xtile0 + (size_t)lrow * C_DIM;   // this lane's row
    const float* pf0    = xtile0 + (size_t)lane * 32;      // 32 lanes x 128B = 4KB tile

    for (int t = 0; t < T_STEPS; ++t) {
        const float* xrow = xrow0 + (size_t)t * xstep;

        // Prefetch the next timestep's contiguous 4KB x tile (one line/lane).
        {
            const int tn = (t + 1 < T_STEPS) ? (t + 1) : t;
            __builtin_prefetch((const void*)(pf0 + (size_t)tn * xstep), 0, 0);
        }

        // ---- A operands for this timestep: x tile (global) + h tile (LDS) ----
        v16h Ax[2], Ah[2];
#pragma unroll
        for (int q = 0; q < 2; ++q) {
            // f16 16x32 A layout: lane holds K runs [c0,c0+8) and [c0+16,c0+24)
            const int c0 = q * 32 + 8 * lhalf;
            v4f f0 = *(const v4f*)(xrow + c0);
            v4f f1 = *(const v4f*)(xrow + c0 + 4);
            v4f f2 = *(const v4f*)(xrow + c0 + 16);
            v4f f3 = *(const v4f*)(xrow + c0 + 20);
            v16h a;
#pragma unroll
            for (int e = 0; e < 4; ++e) {
                a[e]      = (_Float16)f0[e];
                a[4 + e]  = (_Float16)f1[e];
                a[8 + e]  = (_Float16)f2[e];
                a[12 + e] = (_Float16)f3[e];
            }
            Ax[q] = a;

            const _Float16* hr = hbuf + lrow * WPAD + c0;
            v8h h0 = *(const v8h*)(hr);
            v8h h1 = *(const v8h*)(hr + 16);
            v16h ah;
#pragma unroll
            for (int e = 0; e < 8; ++e) { ah[e] = h0[e]; ah[8 + e] = h1[e]; }
            Ah[q] = ah;
        }

        // ---- Per 16-wide output column group: 12 WMMAs + gate nonlinearity ----
#pragma unroll
        for (int i = 0; i < 4; ++i) {
            v8f ar, az, ain, ahn;
#pragma unroll
            for (int e = 0; e < 8; ++e) {
                ar[e] = br[i]; az[e] = bz[i]; ain[e] = bin[i]; ahn[e] = bhn[i];
            }
#pragma unroll
            for (int q = 0; q < 2; ++q) {
                ar  = wmma_f16(Ax[q], Wih[i][0][q], ar);
                ar  = wmma_f16(Ah[q], Whh[i][0][q], ar);
                az  = wmma_f16(Ax[q], Wih[i][1][q], az);
                az  = wmma_f16(Ah[q], Whh[i][1][q], az);
                ain = wmma_f16(Ax[q], Wih[i][2][q], ain);
                ahn = wmma_f16(Ah[q], Whh[i][2][q], ahn);
            }
            v8f hnew;
#pragma unroll
            for (int e = 0; e < 8; ++e) {
                const float r = fast_sigmoid(ar[e]);
                const float z = fast_sigmoid(az[e]);
                const float n = fast_tanh(ain[e] + r * ahn[e]);
                hnew[e] = (1.0f - z) * n + z * hD[i][e];
            }
            hD[i] = hnew;
            // Scatter h_new (f16) back to the wave-private LDS h buffer; the
            // A-layout copy for THIS timestep is already in registers.
#pragma unroll
            for (int j = 0; j < 8; ++j) {
                const int M = j + 8 * lhalf;             // D-layout row
                hbuf[M * WPAD + i * 16 + lrow] = (_Float16)hnew[j];
            }
        }
        // Same-wave DS store->load ordering; pin instruction scheduling.
        __builtin_amdgcn_wave_barrier();
    }

    // Final h -> out[b, s, h] (flat: n*64 + col), f32.
    float* orow = out + (size_t)n0 * H_DIM;
#pragma unroll
    for (int i = 0; i < 4; ++i) {
#pragma unroll
        for (int j = 0; j < 8; ++j) {
            const int M = j + 8 * lhalf;
            orow[M * H_DIM + i * 16 + lrow] = hD[i][j];
        }
    }
}

extern "C" void kernel_launch(void* const* d_in, const int* in_sizes, int n_in,
                              void* d_out, int out_size, void* d_ws, size_t ws_size,
                              hipStream_t stream) {
    const float* chars = (const float*)d_in[0];
    const float* W_ih  = (const float*)d_in[1];
    const float* W_hh  = (const float*)d_in[2];
    const float* b_ih  = (const float*)d_in[3];
    const float* b_hh  = (const float*)d_in[4];
    float* out = (float*)d_out;

    const dim3 grid(NTILES / WAVES);   // 500 blocks
    const dim3 block(WAVES * 32);      // 128 threads = 4 wave32 waves
    gru_scan_kernel<<<grid, block, 0, stream>>>(chars, W_ih, W_hh, b_ih, b_hh, out);
}
